// RRPlus_M3_48515950576135
// MI455X (gfx1250) — matrix operand order repaired and checked by reading, not compiled, not run on hardware
//
#include <hip/hip_runtime.h>
#include <hip/hip_bf16.h>

// ---- CDNA5 WMMA fragment types (wave32) ----
typedef __attribute__((ext_vector_type(16))) __bf16 v16bf;
typedef __attribute__((ext_vector_type(8)))  float  v8f;

// Problem constants
#define B_    4
#define CG    150
#define CGP   160      // Co/Ci padded to multiple of 16
#define NH    9
#define XLEN  32000
#define K1_   79
#define K1P   96       // 79 padded -> 3 k-steps of 32
#define KG    1440     // 160*9 padded gconv reduction -> 45 k-steps
#define X2    2000     // after two pools of 4
#define HOUT  7
// padded lifting input row: 39*256 left zeros, 56*256(+16) right zeros
#define LPAD1 9984
#define XROW  56336    // 9984 + 32000 + 14352
// padded/transposed s2: [b][h][x: -64..2063][ci:160]
#define X2P   2128
#define XOFF  64

// ---------------- small helper kernels ----------------
__global__ void k_init(float* __restrict__ sums1, float* __restrict__ sums2) {
    int t = threadIdx.x;
    if (t < 320) { sums1[t] = 0.f; sums2[t] = 0.f; }
}

// x [4,32000] fp32 -> zero-padded bf16 rows [4][XROW]
__global__ void k_prep_x(const float* __restrict__ x, __bf16* __restrict__ xbp) {
    int idx = blockIdx.x * 256 + threadIdx.x;
    if (idx >= B_ * XROW) return;
    int b = idx / XROW, pos = idx % XROW;
    int xi = pos - LPAD1;
    float v = (xi >= 0 && xi < XLEN) ? x[b * XLEN + xi] : 0.f;
    xbp[idx] = (__bf16)v;
}

// w1 [150,1,79] -> w1b [160][96] (zero padded)
__global__ void k_cvt_w1(const float* __restrict__ w1, __bf16* __restrict__ w1b) {
    int i = blockIdx.x * 256 + threadIdx.x;
    if (i < CGP * K1P) {
        int co = i / K1P, k = i % K1P;
        float v = (co < CG && k < K1_) ? w1[co * K1_ + k] : 0.f;
        w1b[i] = (__bf16)v;
    }
}

// w2 [150,150,3,3] -> w2b [160][1440], kidx = (j*3+k)*160 + ci  ((j,k)-major)
__global__ void k_cvt_w2(const float* __restrict__ w2, __bf16* __restrict__ w2b) {
    int i = blockIdx.x * 256 + threadIdx.x;
    if (i < CGP * KG) {
        int co = i / KG, kidx = i % KG;
        int g = kidx / CGP, ci = kidx % CGP;
        int j = g / 3, kk = g % 3;
        float v = (co < CG && ci < CG) ? w2[((co * CG + ci) * 3 + j) * 3 + kk] : 0.f;
        w2b[i] = (__bf16)v;
    }
}

// mean/var -> scale/shift (training-mode BN, biased var, eps = 2e-5)
__global__ void k_bnparam(const float* __restrict__ sums, const float* __restrict__ g,
                          const float* __restrict__ bet, float* __restrict__ bn, float count) {
    int t = threadIdx.x;
    if (t < CG) {
        float mean = sums[t] / count;
        float var  = sums[160 + t] / count - mean * mean;
        float sc   = g[t] * rsqrtf(var + 2e-5f);
        bn[t] = sc;
        bn[160 + t] = bet[t] - mean * sc;
    }
}

// ---------------- lifting conv (WMMA) + pool1 + BN1 stats ----------------
// grid (250, 9, 4), block 256 (8 waves). Wave w handles pre-pool x-tile of 16.
// GEMM: out[co, x] = sum_k w1[co,k] * x[b, x + d*(k-39)],  D scaled by 1/d.
// Zero-padded xbp makes every gather unconditional (k>=79 killed by zero A rows).
__global__ __launch_bounds__(256) void k_lift(const __bf16* __restrict__ xbp,
                                              const __bf16* __restrict__ w1b,
                                              __bf16* __restrict__ s1,
                                              float* __restrict__ sums1) {
    __shared__ float chSum[CGP], chSq[CGP];
    __shared__ float tile[8][16][17];
    int t = threadIdx.x;
    if (t < CGP) { chSum[t] = 0.f; chSq[t] = 0.f; }
    __syncthreads();

    int wid = t >> 5, lane = t & 31;
    int n = lane & 15, hi = lane >> 4;
    int i = blockIdx.y, b = blockIdx.z;
    int d = 1 << i;
    float invd = 1.0f / (float)d;
    int xg = blockIdx.x * 128 + wid * 16 + n;        // pre-pool x of this lane's column
    const __bf16* xrow = xbp + b * XROW;
    int idx0 = LPAD1 + xg - 39 * d;                  // >= 0 by construction

    v8f acc[10];
    v8f zero = {0.f,0.f,0.f,0.f,0.f,0.f,0.f,0.f};
    #pragma unroll
    for (int c = 0; c < 10; ++c) acc[c] = zero;

    #pragma unroll
    for (int ks = 0; ks < 3; ++ks) {
        // B fragment: lane holds column xg, K = ks*32 + hi*16 + e (no bounds checks)
        v16bf bfrag;
        int kbase = ks * 32 + hi * 16;
        #pragma unroll
        for (int e = 0; e < 16; ++e)
            bfrag[e] = xrow[idx0 + d * (kbase + e)];
        #pragma unroll
        for (int cot = 0; cot < 10; ++cot) {
            // A fragment: row = cot*16 + n, K = ks*32 + hi*16 + e (matches B)
            const __bf16* arow = w1b + (cot * 16 + n) * K1P + ks * 32 + 16 * hi;
            v16bf a;
            #pragma unroll
            for (int e = 0; e < 16; ++e) a[e] = arow[e];
            acc[cot] = __builtin_amdgcn_wmma_f32_16x16x32_bf16(
                false, a, false, bfrag, (short)0, acc[cot], false, false);
        }
    }

    // epilogue per co-tile: spill D to LDS, pool 4->1, store bf16 + BN stats
    for (int cot = 0; cot < 10; ++cot) {
        #pragma unroll
        for (int r = 0; r < 8; ++r) tile[wid][r + 8 * hi][n] = acc[cot][r] * invd;
        __syncthreads();
        #pragma unroll
        for (int h = 0; h < 2; ++h) {
            int p  = lane + 32 * h;                 // 0..63 pooled outputs of this wave
            int m  = p >> 2, px = p & 3;
            float v = tile[wid][m][px * 4];
            v = fmaxf(v, tile[wid][m][px * 4 + 1]);
            v = fmaxf(v, tile[wid][m][px * 4 + 2]);
            v = fmaxf(v, tile[wid][m][px * 4 + 3]);
            int co = cot * 16 + m;
            if (co < CG) {
                int xp = (blockIdx.x * 128 + wid * 16) / 4 + px;
                s1[((b * CG + co) * NH + i) * 8000 + xp] = (__bf16)v;
                atomicAdd(&chSum[co], v);
                atomicAdd(&chSq[co],  v * v);
            }
        }
        __syncthreads();
    }
    if (t < CG) {
        atomicAdd(&sums1[t],        chSum[t]);
        atomicAdd(&sums1[160 + t],  chSq[t]);
    }
}

// -------- BN1 + relu + pool2 + transpose: s1 [4,150,9,8000] -> s2p [4][9][2128][160] ----
// ci is the fastest dim (padded to 160, zero-filled) and x gets a +-64 zero halo,
// so the gconv B fragment is 32 contiguous aligned bytes with no masking.
__global__ __launch_bounds__(256) void k_pool2(const __bf16* __restrict__ s1,
                                               const float* __restrict__ bn1,
                                               __bf16* __restrict__ s2p) {
    int idx = blockIdx.x * 256 + threadIdx.x;
    if (idx >= B_ * NH * X2P * CGP) return;
    int ci = idx % CGP;
    int xp = (idx / CGP) % X2P;
    int h  = (idx / (CGP * X2P)) % NH;
    int b  = idx / (CGP * X2P * NH);
    int x  = xp - XOFF;
    float m = 0.f;
    if (ci < CG && x >= 0 && x < X2) {
        const __bf16* p = s1 + ((size_t)((b * CG + ci) * NH + h) * 8000 + x * 4);
        float sc = bn1[ci], sh = bn1[160 + ci];
        #pragma unroll
        for (int tt = 0; tt < 4; ++tt)
            m = fmaxf(m, fmaxf((float)p[tt] * sc + sh, 0.f));
    }
    s2p[idx] = (__bf16)m;
}

// ---------------- group conv (WMMA) + BN2 stats ----------------
// grid (25, 7, 4), block 160 (5 waves, 16 x each). K = (j*3+k)*160 + ci, 45 steps of 32.
__global__ __launch_bounds__(160) void k_gconv(const __bf16* __restrict__ s2p,
                                               const __bf16* __restrict__ w2b,
                                               __bf16* __restrict__ s3,
                                               float* __restrict__ sums2) {
    __shared__ float chSum[CGP], chSq[CGP];
    int t = threadIdx.x;
    if (t < CGP) { chSum[t] = 0.f; chSq[t] = 0.f; }
    __syncthreads();

    int wid = t >> 5, lane = t & 31;
    int n = lane & 15, hi = lane >> 4;
    int i = blockIdx.y, b = blockIdx.z;
    int d = 1 << i;
    float invd = 1.0f / (float)d;
    int xg = blockIdx.x * 80 + wid * 16 + n;         // output x in 0..1999

    v8f acc[10];
    v8f zero = {0.f,0.f,0.f,0.f,0.f,0.f,0.f,0.f};
    #pragma unroll
    for (int c = 0; c < 10; ++c) acc[c] = zero;

    #pragma unroll 1
    for (int ks = 0; ks < 45; ++ks) {
        int g   = ks / 5;                  // uniform (j,k) per 32-wide k-step
        int ci0 = (ks % 5) * 32;
        int j   = g / 3, kk = g % 3;
        int xs  = xg + d * (kk - 1) + XOFF;          // always 0..2127
        // B fragment: 16 consecutive bf16 (32B aligned) -> 2x global_load_b128
        size_t base = ((size_t)((b * NH + (i + j)) * X2P) + xs) * CGP + ci0 + hi * 16;
        v16bf bfrag;
        #pragma unroll
        for (int e = 0; e < 16; ++e) bfrag[e] = s2p[base + e];
        #pragma unroll
        for (int cot = 0; cot < 10; ++cot) {
            const __bf16* arow = w2b + (cot * 16 + n) * KG + ks * 32 + 16 * hi;
            v16bf a;
            #pragma unroll
            for (int e = 0; e < 16; ++e) a[e] = arow[e];
            acc[cot] = __builtin_amdgcn_wmma_f32_16x16x32_bf16(
                false, a, false, bfrag, (short)0, acc[cot], false, false);
        }
    }

    #pragma unroll
    for (int cot = 0; cot < 10; ++cot) {
        #pragma unroll
        for (int r = 0; r < 8; ++r) {
            int co = cot * 16 + r + 8 * hi;
            if (co < CG) {
                float v = acc[cot][r] * invd;
                s3[((b * CG + co) * HOUT + i) * X2 + xg] = (__bf16)v;
                atomicAdd(&chSum[co], v);
                atomicAdd(&chSq[co],  v * v);
            }
        }
    }
    __syncthreads();
    if (t < CG) {
        atomicAdd(&sums2[t],       chSum[t]);
        atomicAdd(&sums2[160 + t], chSq[t]);
    }
}

// ---- BN2 + relu + pool(4) + mean over x : one wave per (b,c,i) row -> s4 [4200] ----
__global__ __launch_bounds__(256) void k_red(const __bf16* __restrict__ s3,
                                             const float* __restrict__ bn2,
                                             float* __restrict__ s4) {
    int wid = threadIdx.x >> 5, lane = threadIdx.x & 31;
    int row = blockIdx.x * 8 + wid;                  // 0..4199
    if (row >= B_ * CG * HOUT) return;
    int c = (row / HOUT) % CG;
    float sc = bn2[c], sh = bn2[160 + c];
    const __bf16* p = s3 + (size_t)row * X2;
    float accv = 0.f;
    for (int gidx = lane; gidx < 500; gidx += 32) {
        float m = 0.f;
        #pragma unroll
        for (int tt = 0; tt < 4; ++tt) {
            float v = fmaxf((float)p[gidx * 4 + tt] * sc + sh, 0.f);
            m = fmaxf(m, v);
        }
        accv += m;
    }
    #pragma unroll
    for (int off = 16; off > 0; off >>= 1) accv += __shfl_down(accv, off, 32);
    if (lane == 0) s4[row] = accv * (1.0f / 500.0f);
}

// ---- final gconv (w3, k=1, /d) + max over scale -> out [4,10] fp32 ----
__global__ __launch_bounds__(256) void k_final(const float* __restrict__ w3,
                                               const float* __restrict__ s4,
                                               float* __restrict__ out) {
    __shared__ float vals[200];
    int t = threadIdx.x;
    if (t < 200) {
        int b = t / 50, r = t % 50, co = r / 5, i = r % 5;
        float accv = 0.f;
        for (int ci = 0; ci < CG; ++ci) {
            const float* wp = w3 + (co * CG + ci) * 3;   // [10,150,3,1]
            const float* xp = s4 + (b * CG + ci) * HOUT + i;
            accv += wp[0] * xp[0] + wp[1] * xp[1] + wp[2] * xp[2];
        }
        vals[t] = accv / (float)(1 << i);
    }
    __syncthreads();
    if (t < 40) {
        int b = t / 10, co = t % 10;
        float m = vals[b * 50 + co * 5];
        #pragma unroll
        for (int i = 1; i < 5; ++i) m = fmaxf(m, vals[b * 50 + co * 5 + i]);
        out[b * 10 + co] = m;
    }
}

// ---------------- host launcher ----------------
extern "C" void kernel_launch(void* const* d_in, const int* in_sizes, int n_in,
                              void* d_out, int out_size, void* d_ws, size_t ws_size,
                              hipStream_t stream) {
    (void)in_sizes; (void)n_in; (void)out_size; (void)ws_size;
    const float* x  = (const float*)d_in[0];
    const float* w1 = (const float*)d_in[1];
    const float* w2 = (const float*)d_in[2];
    const float* w3 = (const float*)d_in[3];
    const float* g1 = (const float*)d_in[4];
    const float* b1 = (const float*)d_in[5];
    const float* g2 = (const float*)d_in[6];
    const float* b2 = (const float*)d_in[7];
    float* out = (float*)d_out;

    char* ws = (char*)d_ws;
    size_t off = 0;
    auto take = [&](size_t bytes) -> char* {
        char* p = ws + off;
        off = (off + bytes + 255) & ~(size_t)255;
        return p;
    };
    __bf16* xbp   = (__bf16*)take((size_t)B_ * XROW * 2);
    __bf16* w1b   = (__bf16*)take((size_t)CGP * K1P * 2);
    __bf16* w2b   = (__bf16*)take((size_t)CGP * KG * 2);
    float*  sums1 = (float*) take(320 * 4);
    float*  sums2 = (float*) take(320 * 4);
    float*  bn1   = (float*) take(320 * 4);
    float*  bn2   = (float*) take(320 * 4);
    float*  s4    = (float*) take((size_t)B_ * CG * HOUT * 4);
    __bf16* s1    = (__bf16*)take((size_t)B_ * CG * NH * 8000 * 2);
    __bf16* s2p   = (__bf16*)take((size_t)B_ * NH * X2P * CGP * 2);
    __bf16* s3    = (__bf16*)take((size_t)B_ * CG * HOUT * X2 * 2);

    k_init   <<<1, 320, 0, stream>>>(sums1, sums2);
    k_prep_x <<<(B_ * XROW + 255) / 256, 256, 0, stream>>>(x, xbp);
    k_cvt_w1 <<<(CGP * K1P + 255) / 256, 256, 0, stream>>>(w1, w1b);
    k_cvt_w2 <<<(CGP * KG + 255) / 256, 256, 0, stream>>>(w2, w2b);

    k_lift   <<<dim3(250, NH, B_), 256, 0, stream>>>(xbp, w1b, s1, sums1);
    k_bnparam<<<1, 256, 0, stream>>>(sums1, g1, b1, bn1, (float)(B_ * NH * 8000));
    k_pool2  <<<(B_ * NH * X2P * CGP + 255) / 256, 256, 0, stream>>>(s1, bn1, s2p);

    k_gconv  <<<dim3(25, HOUT, B_), 160, 0, stream>>>(s2p, w2b, s3, sums2);
    k_bnparam<<<1, 256, 0, stream>>>(sums2, g2, b2, bn2, (float)(B_ * HOUT * X2));

    k_red    <<<(B_ * CG * HOUT + 7) / 8, 256, 0, stream>>>(s3, bn2, s4);
    k_final  <<<1, 256, 0, stream>>>(w3, s4, out);
}
